// GATv2_46420006536041
// MI455X (gfx1250) — compile-verified, matrix operand order checked by reading, not run on hardware
//
#include <hip/hip_runtime.h>
#include <hip/hip_bf16.h>

#define N_NODES   1536
#define IN_F      1433
#define KP        1440     // IN_F padded to multiple of 32 for WMMA K-loop
#define NH        64       // hidden
#define HEADS     8
#define FPH       8        // features per head
#define NC        7        // classes
#define NEG_SLOPE 0.2f

typedef __attribute__((ext_vector_type(16))) _Float16 v16h;
typedef __attribute__((ext_vector_type(8)))  _Float16 v8h;
typedef __attribute__((ext_vector_type(8)))  float    v8f;

// ---------------------------------------------------------------------------
// f32 -> f16 conversion with K padding. Xh is (N_NODES x KP) row-major.
__global__ void cvt_x_kernel(const float* __restrict__ x, _Float16* __restrict__ xh) {
  int idx = blockIdx.x * blockDim.x + threadIdx.x;
  if (idx >= N_NODES * KP) return;
  int m = idx / KP, k = idx % KP;
  xh[idx] = (k < IN_F) ? (_Float16)x[m * IN_F + k] : (_Float16)0.0f;
}

// W1 (IN_F x NH) -> W1T f16 (NH x KP) so B-fragment loads are K-contiguous.
__global__ void cvt_w1t_kernel(const float* __restrict__ w1, _Float16* __restrict__ w1t) {
  int idx = blockIdx.x * blockDim.x + threadIdx.x;
  if (idx >= NH * KP) return;
  int n = idx / KP, k = idx % KP;
  w1t[idx] = (k < IN_F) ? (_Float16)w1[k * NH + n] : (_Float16)0.0f;
}

// ---------------------------------------------------------------------------
// g1 = Xh @ W1  via v_wmma_f32_16x16x32_f16.
// Block = 128 threads (4 waves). Wave w computes the 16x16 tile at
// (m0 = blockIdx.x*16, n0 = w*16). K-loop: 45 iterations of K=32.
// CDNA5 16-bit A fragment layout: lane l -> row (l&15), K-base (l>>4)*8;
// halves 0..7 = K..K+7, halves 8..15 = K+16..K+23. B mirrors A via W1T.
__global__ __launch_bounds__(128) void gemm1_wmma_kernel(
    const _Float16* __restrict__ A,   // (N_NODES x KP)
    const _Float16* __restrict__ BT,  // (NH x KP)  == W1^T
    float* __restrict__ C) {          // (N_NODES x NH)
  const int lane = threadIdx.x & 31;
  const int wave = threadIdx.x >> 5;
  const int m0   = blockIdx.x * 16;
  const int n0   = wave * 16;
  const int r    = lane & 15;
  const int kb   = (lane >> 4) * 8;

  const _Float16* __restrict__ arow = A  + (size_t)(m0 + r) * KP + kb;
  const _Float16* __restrict__ brow = BT + (size_t)(n0 + r) * KP + kb;

  v8f c = {};
  for (int k0 = 0; k0 < KP; k0 += 32) {
    v8h a_lo = *(const v8h*)(arow + k0);
    v8h a_hi = *(const v8h*)(arow + k0 + 16);
    v8h b_lo = *(const v8h*)(brow + k0);
    v8h b_hi = *(const v8h*)(brow + k0 + 16);
    v16h a, b;
#pragma unroll
    for (int t = 0; t < 8; ++t) {
      a[t] = a_lo[t]; a[t + 8] = a_hi[t];
      b[t] = b_lo[t]; b[t + 8] = b_hi[t];
    }
    // 8 args: (neg_a, A, neg_b, B, c_mod, C, reuse_a, reuse_b)
    c = __builtin_amdgcn_wmma_f32_16x16x32_f16(false, a, false, b,
                                               (short)0, c, false, false);
  }
  // C/D layout: VGPR v -> M = v (lanes 0-15) or v+8 (lanes 16-31); N = lane&15
  const int rofs = (lane >> 4) * 8;
  const int col  = lane & 15;
#pragma unroll
  for (int v = 0; v < 8; ++v)
    C[(size_t)(m0 + v + rofs) * NH + n0 + col] = c[v];
}

// ---------------------------------------------------------------------------
// Layer-1 GATv2 attention + ELU. One block (256 thr = 8 waves) per node i.
// Two-pass softmax with the full masked logit row resident in LDS (48 KB).
__global__ __launch_bounds__(256) void attn1_kernel(
    const float* __restrict__ g,            // (N_NODES x NH) = g1
    const unsigned char* __restrict__ adj,  // (N_NODES x N_NODES) bool
    const float* __restrict__ a1,           // (FPH)
    float* __restrict__ h1) {               // (N_NODES x NH), post-ELU
  const int i   = blockIdx.x;
  const int tid = threadIdx.x;

  __shared__ float gi[NH];
  __shared__ float av[FPH];
  __shared__ float e_s[N_NODES * HEADS];  // 48 KB logits
  __shared__ float redbuf[256 * HEADS];   // 8 KB max-reduction scratch
  __shared__ float mx[HEADS];
  __shared__ float s_sum[HEADS];
  __shared__ float s_acc[NH];

  const float NEG_INF = -__builtin_inff();

  if (tid < NH)    gi[tid]    = g[(size_t)i * NH + tid];
  if (tid < FPH)   av[tid]    = a1[tid];
  if (tid < HEADS) s_sum[tid] = 0.0f;
  if (tid < NH)    s_acc[tid] = 0.0f;
  __syncthreads();

  // Pass 1: logits + per-thread running max per head.
  float lmax[HEADS];
#pragma unroll
  for (int h = 0; h < HEADS; ++h) lmax[h] = NEG_INF;

  for (int j = tid; j < N_NODES; j += 256) {
    const bool conn = adj[(size_t)i * N_NODES + j] != 0;
#pragma unroll
    for (int h = 0; h < HEADS; ++h) {
      float e = NEG_INF;
      if (conn) {
        e = 0.0f;
#pragma unroll
        for (int f = 0; f < FPH; ++f) {
          float s = gi[h * FPH + f] + g[(size_t)j * NH + h * FPH + f];
          s = (s > 0.0f) ? s : NEG_SLOPE * s;
          e = fmaf(av[f], s, e);
        }
        lmax[h] = fmaxf(lmax[h], e);
      }
      e_s[j * HEADS + h] = e;
    }
  }
#pragma unroll
  for (int h = 0; h < HEADS; ++h) redbuf[tid * HEADS + h] = lmax[h];
  __syncthreads();
  if (tid < HEADS) {
    float m = NEG_INF;
    for (int t = 0; t < 256; ++t) m = fmaxf(m, redbuf[t * HEADS + tid]);
    mx[tid] = m;
  }
  __syncthreads();

  // Pass 2: exp, denominator, weighted aggregation of g[j].
  float acc[NH];
  float ssum[HEADS];
#pragma unroll
  for (int c = 0; c < NH; ++c) acc[c] = 0.0f;
#pragma unroll
  for (int h = 0; h < HEADS; ++h) ssum[h] = 0.0f;

  for (int j = tid; j < N_NODES; j += 256) {
#pragma unroll
    for (int h = 0; h < HEADS; ++h) {
      float e = e_s[j * HEADS + h];
      if (e > NEG_INF) {
        float w = __expf(e - mx[h]);
        ssum[h] += w;
#pragma unroll
        for (int f = 0; f < FPH; ++f)
          acc[h * FPH + f] = fmaf(w, g[(size_t)j * NH + h * FPH + f], acc[h * FPH + f]);
      }
    }
  }
#pragma unroll
  for (int h = 0; h < HEADS; ++h) atomicAdd(&s_sum[h], ssum[h]);
#pragma unroll
  for (int c = 0; c < NH; ++c) atomicAdd(&s_acc[c], acc[c]);
  __syncthreads();

  if (tid < NH) {
    float o = s_acc[tid] / s_sum[tid / FPH];
    h1[(size_t)i * NH + tid] = (o > 0.0f) ? o : (__expf(o) - 1.0f);  // ELU
  }
}

// ---------------------------------------------------------------------------
// g2 = h1 @ W2   (1536x64 @ 64x7) — tiny, scalar.
__global__ void gemm2_kernel(const float* __restrict__ h1,
                             const float* __restrict__ W2,
                             float* __restrict__ g2) {
  int idx = blockIdx.x * blockDim.x + threadIdx.x;
  if (idx >= N_NODES * NC) return;
  int i = idx / NC, c = idx % NC;
  float s = 0.0f;
#pragma unroll 8
  for (int k = 0; k < NH; ++k) s = fmaf(h1[(size_t)i * NH + k], W2[k * NC + c], s);
  g2[idx] = s;
}

// ---------------------------------------------------------------------------
// Layer-2 attention, single head, NC features. Mean over 1 head == identity.
__global__ __launch_bounds__(256) void attn2_kernel(
    const float* __restrict__ g2,           // (N_NODES x NC)
    const unsigned char* __restrict__ adj,
    const float* __restrict__ a2,           // (NC)
    float* __restrict__ out) {              // (N_NODES x NC)
  const int i   = blockIdx.x;
  const int tid = threadIdx.x;

  __shared__ float gi[NC];
  __shared__ float av[NC];
  __shared__ float e_s[N_NODES];   // 6 KB
  __shared__ float redbuf[256];
  __shared__ float s_acc[NC];
  __shared__ float mxs;
  __shared__ float ssum_s;

  const float NEG_INF = -__builtin_inff();

  if (tid < NC) { gi[tid] = g2[(size_t)i * NC + tid]; av[tid] = a2[tid]; s_acc[tid] = 0.0f; }
  if (tid == 0) ssum_s = 0.0f;
  __syncthreads();

  float lmax = NEG_INF;
  for (int j = tid; j < N_NODES; j += 256) {
    float e = NEG_INF;
    if (adj[(size_t)i * N_NODES + j] != 0) {
      e = 0.0f;
#pragma unroll
      for (int c = 0; c < NC; ++c) {
        float s = gi[c] + g2[(size_t)j * NC + c];
        s = (s > 0.0f) ? s : NEG_SLOPE * s;
        e = fmaf(av[c], s, e);
      }
      lmax = fmaxf(lmax, e);
    }
    e_s[j] = e;
  }
  redbuf[tid] = lmax;
  __syncthreads();
  if (tid == 0) {
    float m = NEG_INF;
    for (int t = 0; t < 256; ++t) m = fmaxf(m, redbuf[t]);
    mxs = m;
  }
  __syncthreads();

  float acc[NC];
#pragma unroll
  for (int c = 0; c < NC; ++c) acc[c] = 0.0f;
  float ssum = 0.0f;

  for (int j = tid; j < N_NODES; j += 256) {
    float e = e_s[j];
    if (e > NEG_INF) {
      float w = __expf(e - mxs);
      ssum += w;
#pragma unroll
      for (int c = 0; c < NC; ++c) acc[c] = fmaf(w, g2[(size_t)j * NC + c], acc[c]);
    }
  }
  atomicAdd(&ssum_s, ssum);
#pragma unroll
  for (int c = 0; c < NC; ++c) atomicAdd(&s_acc[c], acc[c]);
  __syncthreads();

  if (tid < NC) out[(size_t)i * NC + tid] = s_acc[tid] / ssum_s;
}

// ---------------------------------------------------------------------------
extern "C" void kernel_launch(void* const* d_in, const int* in_sizes, int n_in,
                              void* d_out, int out_size, void* d_ws, size_t ws_size,
                              hipStream_t stream) {
  (void)in_sizes; (void)n_in; (void)out_size; (void)ws_size;
  // setup_inputs order: x, adj_mat, W1, a1, W2, a2
  const float*         x   = (const float*)d_in[0];
  const unsigned char* adj = (const unsigned char*)d_in[1];  // jax bool = 1 byte
  const float*         W1  = (const float*)d_in[2];
  const float*         a1  = (const float*)d_in[3];
  const float*         W2  = (const float*)d_in[4];
  const float*         a2  = (const float*)d_in[5];
  float* out = (float*)d_out;

  // Workspace layout (all 256B-aligned offsets)
  char* ws = (char*)d_ws;
  _Float16* xh  = (_Float16*)(ws + 0);            // 1536*1440*2 = 4,423,680
  _Float16* w1t = (_Float16*)(ws + 4423680);      //   64*1440*2 =   184,320
  float*    g1  = (float*)   (ws + 4608000);      //   1536*64*4 =   393,216
  float*    h1  = (float*)   (ws + 5001216);      //   1536*64*4 =   393,216
  float*    g2  = (float*)   (ws + 5394432);      //    1536*7*4 =    43,008

  // 1) f16 conversion (+ K padding / W1 transpose)
  cvt_x_kernel  <<<(N_NODES * KP + 255) / 256, 256, 0, stream>>>(x, xh);
  cvt_w1t_kernel<<<(NH * KP + 255) / 256, 256, 0, stream>>>(W1, w1t);

  // 2) WMMA projection GEMM: g1 = Xh @ W1
  gemm1_wmma_kernel<<<N_NODES / 16, 128, 0, stream>>>(xh, w1t, g1);

  // 3) Layer-1 attention + ELU
  attn1_kernel<<<N_NODES, 256, 0, stream>>>(g1, adj, a1, h1);

  // 4) Layer-2 projection (tiny)
  gemm2_kernel<<<(N_NODES * NC + 255) / 256, 256, 0, stream>>>(h1, W2, g2);

  // 5) Layer-2 attention -> output
  attn2_kernel<<<N_NODES, 256, 0, stream>>>(g2, adj, a2, out);
}
